// FeatureRetrievalAssociativeLM_34368328303145
// MI455X (gfx1250) — compile-verified, matrix operand order checked by reading, not run on hardware
//
#include <hip/hip_runtime.h>
#include <hip/hip_fp16.h>
#include <cstdint>

typedef _Float16 half_t;
typedef __attribute__((ext_vector_type(16))) _Float16 v16h;
typedef __attribute__((ext_vector_type(8)))  _Float16 v8h;
typedef __attribute__((ext_vector_type(8)))  float    v8f;

static constexpr int Vv = 32000, Ee = 512, Hh = 1024, Mm = 256;
static constexpr int Bb = 2, Tt = 2048, STRIDEc = 4, Ss = Tt / STRIDEc; // 512
static constexpr int BT = Bb * Tt; // 4096

union FragU { v16h v; v8h h[2]; };

// ---------------------------------------------------------------------------
// fp32 -> fp16 convert
// ---------------------------------------------------------------------------
__global__ void f32_to_f16_kernel(const float* __restrict__ src,
                                  half_t* __restrict__ dst, long long n) {
  long long i = (long long)blockIdx.x * blockDim.x + threadIdx.x;
  if (i < n) dst[i] = (half_t)src[i];
}

// ---------------------------------------------------------------------------
// Embedding gather -> fp16
// ---------------------------------------------------------------------------
__global__ void embed_kernel(const int* __restrict__ ids,
                             const float* __restrict__ embw,
                             half_t* __restrict__ emb16) {
  const int tok = blockIdx.x;              // 0..BT-1
  const int id = ids[tok];
  const float* src = embw + (size_t)id * Ee;
  half_t* dst = emb16 + (size_t)tok * Ee;
  for (int e = threadIdx.x; e < Ee; e += blockDim.x) dst[e] = (half_t)src[e];
}

// ---------------------------------------------------------------------------
// Generic batched WMMA GEMM:  C[m,n] = scale * sum_k A[m,k]*B[n,k]  (+ bias[n])
// A: fp16, row stride lda (elements), batch stride strideA
// B: fp16, row stride ldb, batch stride strideB   (B is the (N,K) weight)
// Epilogue: EPI=0 -> fp32 store; EPI=1 -> fp16 store; EPI=2 -> relu^2, fp16
// Block tile 128x128, K-step 32, 256 threads = 8 wave32 (2x4), wave tile 64x32.
// All problem dims are multiples of the tile sizes (checked host-side).
// ---------------------------------------------------------------------------
template <int EPI>
__global__ __launch_bounds__(256)
void wmma_gemm_kernel(const half_t* __restrict__ A, int lda, long long strideA,
                      const half_t* __restrict__ Bm, int ldb, long long strideB,
                      void* __restrict__ Cout, int ldc, long long strideC,
                      const float* __restrict__ bias, float scale, int Ktot) {
  __shared__ half_t As[128][40];   // 32 halves + 8 pad; row stride 80B (16B-aligned)
  __shared__ half_t Bs[128][40];

  const int tid  = threadIdx.x;
  const int lane = tid & 31;
  const int wave = tid >> 5;
  const int wm   = wave >> 2;      // 0..1
  const int wn   = wave & 3;       // 0..3

  A  += (long long)blockIdx.z * strideA;
  Bm += (long long)blockIdx.z * strideB;
  const long long cb = (long long)blockIdx.z * strideC;

  const long long arow0 = (long long)blockIdx.x * 128;
  const long long brow0 = (long long)blockIdx.y * 128;

  const int lrow = tid >> 2;       // 0..63
  const int lcol = (tid & 3) * 8;  // 0,8,16,24

  v8f acc[4][2] = {};

  for (int k0 = 0; k0 < Ktot; k0 += 32) {
#pragma unroll
    for (int rr = 0; rr < 128; rr += 64) {
      const int r = lrow + rr;
      *(v8h*)&As[r][lcol] = *(const v8h*)(A  + (arow0 + r) * (long long)lda + k0 + lcol);
      *(v8h*)&Bs[r][lcol] = *(const v8h*)(Bm + (brow0 + r) * (long long)ldb + k0 + lcol);
    }
    __syncthreads();

    const int lr = lane & 15;
    const int kb = (lane & 16) ? 8 : 0;   // ISA 16-bit A layout: hi lanes hold K+8
    v16h af[4], bf[2];
#pragma unroll
    for (int i = 0; i < 4; ++i) {
      FragU f;
      const int r = wm * 64 + i * 16 + lr;
      f.h[0] = *(const v8h*)&As[r][kb];        // K = kb .. kb+7
      f.h[1] = *(const v8h*)&As[r][16 + kb];   // K = 16+kb .. 23+kb
      af[i] = f.v;
    }
#pragma unroll
    for (int j = 0; j < 2; ++j) {
      FragU f;
      const int r = wn * 32 + j * 16 + lr;
      f.h[0] = *(const v8h*)&Bs[r][kb];
      f.h[1] = *(const v8h*)&Bs[r][16 + kb];
      bf[j] = f.v;
    }
#pragma unroll
    for (int i = 0; i < 4; ++i)
#pragma unroll
      for (int j = 0; j < 2; ++j)
        acc[i][j] = __builtin_amdgcn_wmma_f32_16x16x32_f16(
            false, af[i], false, bf[j], (short)0, acc[i][j], false, false);
    __syncthreads();
  }

  // Epilogue. D layout: lane<16 -> N=lane, M=vgpr; lane>=16 -> N=lane-16, M=vgpr+8.
  const int lr = lane & 15;
  const int mo = (lane & 16) ? 8 : 0;
#pragma unroll
  for (int i = 0; i < 4; ++i) {
#pragma unroll
    for (int j = 0; j < 2; ++j) {
      const long long n = brow0 + wn * 32 + j * 16 + lr;
      const float bv = bias ? bias[n] : 0.0f;
#pragma unroll
      for (int v = 0; v < 8; ++v) {
        const long long m = arow0 + wm * 64 + i * 16 + mo + v;
        const float x = acc[i][j][v] * scale + bv;
        const long long off = cb + m * (long long)ldc + n;
        if (EPI == 0) {
          ((float*)Cout)[off] = x;
        } else if (EPI == 1) {
          ((half_t*)Cout)[off] = (half_t)x;
        } else {
          const float r = x > 0.0f ? x : 0.0f;
          ((half_t*)Cout)[off] = (half_t)(r * r);
        }
      }
    }
  }
}

// ---------------------------------------------------------------------------
// GRU recurrence: one workgroup per batch element, h kept in LDS, w_hh in L2.
// Gate layout in xg / w_hh / b_hh: [0:H)=r, [H:2H)=z, [2H:3H)=n.
// ---------------------------------------------------------------------------
__global__ __launch_bounds__(1024)
void gru_scan_kernel(const float* __restrict__ xg, const float* __restrict__ w_hh,
                     const float* __restrict__ b_hh, half_t* __restrict__ states16) {
  __shared__ float hs[Hh];
  const int b = blockIdx.x;
  const int j = threadIdx.x;
  hs[j] = 0.0f;
  __syncthreads();

  const float* wr = w_hh + (size_t)j * Hh;
  const float* wz = w_hh + (size_t)(j + Hh) * Hh;
  const float* wn = w_hh + (size_t)(j + 2 * Hh) * Hh;
  const float bhr = b_hh[j], bhz = b_hh[j + Hh], bhn = b_hh[j + 2 * Hh];

  for (int t = 0; t < Tt; ++t) {
    float gr = bhr, gz = bhz, gn = bhn;
#pragma unroll 4
    for (int k = 0; k < Hh; k += 4) {
      const float4 hv = *(const float4*)&hs[k];
      const float4 a = *(const float4*)&wr[k];
      const float4 c = *(const float4*)&wz[k];
      const float4 d = *(const float4*)&wn[k];
      gr = fmaf(a.x, hv.x, fmaf(a.y, hv.y, fmaf(a.z, hv.z, fmaf(a.w, hv.w, gr))));
      gz = fmaf(c.x, hv.x, fmaf(c.y, hv.y, fmaf(c.z, hv.z, fmaf(c.w, hv.w, gz))));
      gn = fmaf(d.x, hv.x, fmaf(d.y, hv.y, fmaf(d.z, hv.z, fmaf(d.w, hv.w, gn))));
    }
    const float* gi = xg + ((size_t)b * Tt + t) * (3 * Hh);
    const float r   = 1.0f / (1.0f + expf(-(gi[j] + gr)));
    const float z   = 1.0f / (1.0f + expf(-(gi[j + Hh] + gz)));
    const float nst = tanhf(gi[j + 2 * Hh] + r * gn);
    const float hnew = (1.0f - z) * nst + z * hs[j];
    __syncthreads();            // all dot-products done reading hs
    hs[j] = hnew;
    __syncthreads();
    states16[((size_t)b * Tt + t) * Hh + j] = (half_t)hnew;
  }
}

// ---------------------------------------------------------------------------
// Masked softmax over scores[b,t,0:S]; valid keys: s*STRIDE < t.
// Rows with no valid key (t==0) produce all-zero attention (matches reference
// after its att*mask renormalization).
// ---------------------------------------------------------------------------
__global__ __launch_bounds__(256)
void softmax_kernel(const float* __restrict__ scores, half_t* __restrict__ att) {
  const int t = blockIdx.x, b = blockIdx.y;
  const float* row = scores + ((size_t)b * Tt + t) * Ss;
  half_t* orow = att + ((size_t)b * Tt + t) * Ss;
  const int nv = (t + STRIDEc - 1) / STRIDEc;   // #s with s*STRIDE < t

  __shared__ float red[256];
  float lmax = -3.4e38f;
  for (int s = threadIdx.x; s < nv; s += 256) lmax = fmaxf(lmax, row[s]);
  red[threadIdx.x] = lmax;
  __syncthreads();
  for (int o = 128; o > 0; o >>= 1) {
    if (threadIdx.x < o) red[threadIdx.x] = fmaxf(red[threadIdx.x], red[threadIdx.x + o]);
    __syncthreads();
  }
  const float m = red[0];
  __syncthreads();

  float lsum = 0.0f;
  for (int s = threadIdx.x; s < nv; s += 256) lsum += expf(row[s] - m);
  red[threadIdx.x] = lsum;
  __syncthreads();
  for (int o = 128; o > 0; o >>= 1) {
    if (threadIdx.x < o) red[threadIdx.x] += red[threadIdx.x + o];
    __syncthreads();
  }
  const float sum = red[0];
  const float inv = sum > 0.0f ? 1.0f / sum : 0.0f;

  for (int s = threadIdx.x; s < Ss; s += 256)
    orow[s] = (half_t)((s < nv) ? expf(row[s] - m) * inv : 0.0f);
}

// ---------------------------------------------------------------------------
// mixed = base + sigmoid(gate_pre) * retrieved   (fp16 out for final GEMM)
// ---------------------------------------------------------------------------
__global__ void mix_kernel(const float* __restrict__ base,
                           const float* __restrict__ gatepre,
                           const float* __restrict__ retr,
                           half_t* __restrict__ mixed, long long n) {
  long long i = (long long)blockIdx.x * blockDim.x + threadIdx.x;
  if (i < n) {
    const float g = 1.0f / (1.0f + expf(-gatepre[i]));
    mixed[i] = (half_t)(base[i] + g * retr[i]);
  }
}

// ---------------------------------------------------------------------------
extern "C" void kernel_launch(void* const* d_in, const int* in_sizes, int n_in,
                              void* d_out, int out_size, void* d_ws, size_t ws_size,
                              hipStream_t stream) {
  (void)in_sizes; (void)n_in; (void)out_size; (void)ws_size;

  const int*   ids     = (const int*)  d_in[0];
  const float* emb_w   = (const float*)d_in[1];
  const float* w_ih    = (const float*)d_in[2];
  const float* w_hh    = (const float*)d_in[3];
  const float* b_ih    = (const float*)d_in[4];
  const float* b_hh    = (const float*)d_in[5];
  const float* q_w     = (const float*)d_in[6];
  const float* q_b     = (const float*)d_in[7];
  const float* k_w     = (const float*)d_in[8];
  const float* k_b     = (const float*)d_in[9];
  const float* hfc_w   = (const float*)d_in[10];
  const float* hfc_b   = (const float*)d_in[11];
  const float* hproj_w = (const float*)d_in[12];
  const float* hproj_b = (const float*)d_in[13];
  const float* v_w     = (const float*)d_in[14];
  const float* gate_w  = (const float*)d_in[15];
  const float* gate_b  = (const float*)d_in[16];
  const float* out_b   = (const float*)d_in[17];
  float* out = (float*)d_out;

  char* ws = (char*)d_ws;
  size_t off = 0;
  auto alloc = [&](size_t bytes) -> void* {
    void* p = ws + off;
    off = (off + bytes + 255) & ~(size_t)255;
    return p;
  };

  half_t* embW16   = (half_t*)alloc((size_t)Vv * Ee * 2);
  half_t* wih16    = (half_t*)alloc((size_t)3 * Hh * Ee * 2);
  half_t* hfcw16   = (half_t*)alloc((size_t)4 * Ee * Hh * 2);
  half_t* hprojw16 = (half_t*)alloc((size_t)Ee * 4 * Ee * 2);
  half_t* qw16     = (half_t*)alloc((size_t)Mm * Hh * 2);
  half_t* kw16     = (half_t*)alloc((size_t)Mm * Hh * 2);
  half_t* vw16     = (half_t*)alloc((size_t)Ee * Ee * 2);
  half_t* gatew16  = (half_t*)alloc((size_t)Ee * Hh * 2);
  half_t* emb16    = (half_t*)alloc((size_t)BT * Ee * 2);
  float*  xg       = (float*) alloc((size_t)BT * 3 * Hh * 4);
  half_t* states16 = (half_t*)alloc((size_t)BT * Hh * 2);
  half_t* head16   = (half_t*)alloc((size_t)BT * 4 * Ee * 2);
  float*  base32   = (float*) alloc((size_t)BT * Ee * 4);
  half_t* qk16     = (half_t*)alloc((size_t)BT * Mm * 2);
  half_t* mk16     = (half_t*)alloc((size_t)Bb * Ss * Mm * 2);
  half_t* srcvT16  = (half_t*)alloc((size_t)Bb * Ee * Ss * 2);
  float*  scores32 = (float*) alloc((size_t)Bb * Tt * Ss * 4);
  half_t* att16    = (half_t*)alloc((size_t)Bb * Tt * Ss * 2);
  float*  retr32   = (float*) alloc((size_t)BT * Ee * 4);
  float*  gate32   = (float*) alloc((size_t)BT * Ee * 4);
  half_t* mixed16  = (half_t*)alloc((size_t)BT * Ee * 2);

  auto conv = [&](const float* s, half_t* d, long long n) {
    f32_to_f16_kernel<<<dim3((unsigned)((n + 255) / 256)), dim3(256), 0, stream>>>(s, d, n);
  };
  conv(emb_w,   embW16,   (long long)Vv * Ee);
  conv(w_ih,    wih16,    (long long)3 * Hh * Ee);
  conv(hfc_w,   hfcw16,   (long long)4 * Ee * Hh);
  conv(hproj_w, hprojw16, (long long)Ee * 4 * Ee);
  conv(q_w,     qw16,     (long long)Mm * Hh);
  conv(k_w,     kw16,     (long long)Mm * Hh);
  conv(v_w,     vw16,     (long long)Ee * Ee);
  conv(gate_w,  gatew16,  (long long)Ee * Hh);

  embed_kernel<<<dim3(BT), dim3(256), 0, stream>>>(ids, emb_w, emb16);

  // xg = emb @ w_ih^T + b_ih : (4096 x 3072), K=512, fp32 out
  wmma_gemm_kernel<0><<<dim3(BT / 128, 3 * Hh / 128, 1), 256, 0, stream>>>(
      emb16, Ee, 0, wih16, Ee, 0, xg, 3 * Hh, 0, b_ih, 1.0f, Ee);

  // GRU scan (sequential over T)
  gru_scan_kernel<<<dim3(Bb), dim3(Hh), 0, stream>>>(xg, w_hh, b_hh, states16);

  // head = relu(states @ hfc_w^T + b)^2 : (4096 x 2048), K=1024, fp16 out
  wmma_gemm_kernel<2><<<dim3(BT / 128, 4 * Ee / 128, 1), 256, 0, stream>>>(
      states16, Hh, 0, hfcw16, Hh, 0, head16, 4 * Ee, 0, hfc_b, 1.0f, Hh);

  // base = head @ hproj_w^T + b : (4096 x 512), K=2048, fp32 out
  wmma_gemm_kernel<0><<<dim3(BT / 128, Ee / 128, 1), 256, 0, stream>>>(
      head16, 4 * Ee, 0, hprojw16, 4 * Ee, 0, base32, Ee, 0, hproj_b, 1.0f, 4 * Ee);

  // qk = states @ q_w^T + q_b : (4096 x 256), K=1024, fp16 out
  wmma_gemm_kernel<1><<<dim3(BT / 128, Mm / 128, 1), 256, 0, stream>>>(
      states16, Hh, 0, qw16, Hh, 0, qk16, Mm, 0, q_b, 1.0f, Hh);

  // mk = src_states @ k_w^T + k_b : batched (B) x (512 x 256), strided-row A
  wmma_gemm_kernel<1><<<dim3(Ss / 128, Mm / 128, Bb), 256, 0, stream>>>(
      states16, STRIDEc * Hh, (long long)Tt * Hh,
      kw16, Hh, 0,
      mk16, Mm, (long long)Ss * Mm, k_b, 1.0f, Hh);

  // srcvT[e,s] = sum_k v_w[e,k] * src_values[s,k] : batched (512 x 512), K=512
  // (computed transposed so the retrieval GEMM stays NT-layout)
  wmma_gemm_kernel<1><<<dim3(Ee / 128, Ss / 128, Bb), 256, 0, stream>>>(
      vw16, Ee, 0,
      emb16, STRIDEc * Ee, (long long)Tt * Ee,
      srcvT16, Ss, (long long)Ee * Ss, nullptr, 1.0f, Ee);

  // scores = qk @ mk^T / sqrt(M) : batched (2048 x 512), K=256, fp32 out
  wmma_gemm_kernel<0><<<dim3(Tt / 128, Ss / 128, Bb), 256, 0, stream>>>(
      qk16, Mm, (long long)Tt * Mm,
      mk16, Mm, (long long)Ss * Mm,
      scores32, Ss, (long long)Tt * Ss, nullptr, 0.0625f, Mm);

  // masked softmax -> att16
  softmax_kernel<<<dim3(Tt, Bb), 256, 0, stream>>>(scores32, att16);

  // retrieved = att @ srcv : batched (2048 x 512), K=S=512, fp32 out
  wmma_gemm_kernel<0><<<dim3(Tt / 128, Ee / 128, Bb), 256, 0, stream>>>(
      att16, Ss, (long long)Tt * Ss,
      srcvT16, Ss, (long long)Ee * Ss,
      retr32, Ee, (long long)Tt * Ee, nullptr, 1.0f, Ss);

  // gate_pre = states @ gate_w^T + b : (4096 x 512), K=1024, fp32 out
  wmma_gemm_kernel<0><<<dim3(BT / 128, Ee / 128, 1), 256, 0, stream>>>(
      states16, Hh, 0, gatew16, Hh, 0, gate32, Ee, 0, gate_b, 1.0f, Hh);

  // mixed = base + sigmoid(gate_pre) * retrieved  (fp16)
  {
    const long long n = (long long)BT * Ee;
    mix_kernel<<<dim3((unsigned)((n + 255) / 256)), 256, 0, stream>>>(
        base32, gate32, retr32, mixed16, n);
  }

  // logits = mixed @ emb_weight^T + out_bias : (4096 x 32000), K=512, fp32 out
  wmma_gemm_kernel<0><<<dim3(BT / 128, Vv / 128, 1), 256, 0, stream>>>(
      mixed16, Ee, 0, embW16, Ee, 0, out, Vv, 0, out_b, 1.0f, Ee);
}
